// MSA_32057635897504
// MI455X (gfx1250) — compile-verified, matrix operand order checked by reading, not run on hardware
//
#include <hip/hip_runtime.h>
#include <hip/hip_bf16.h>

// MI455X (gfx1250) fused relative-position multi-head attention.
// Pipeline:
//   [cvt: fp32 -> bf16 copies of x, wq/wk/wv, dist_emb]  (one-shot, streaming)
//   [qkv_proj: pure-bf16 WMMA GEMM -> q/k/vT in ws]
//   [attn: flash attention; relative-key bias slab computed via WMMA into LDS]
// Workspace (bf16): q[8][12][1024][64], k[..], vT[8][12][64][1024],
//                   xb[8192][768], wb[3][768][768], eb[2047][64]  ~= 54 MB.

typedef __attribute__((ext_vector_type(16))) __bf16 v16bf;
typedef __attribute__((ext_vector_type(8)))  __bf16 v8bf;
typedef __attribute__((ext_vector_type(8)))  float  v8f;

static constexpr int B_  = 8;
static constexpr int N_  = 1024;
static constexpr int D_  = 768;
static constexpr int H_  = 12;
static constexpr int P_  = 64;
static constexpr int EROWS = 2 * 1024 - 1;   // 2047 dist_emb rows
static constexpr float LOG2E = 1.4426950408889634f;

__device__ __forceinline__ v8f wmma_bf16(v16bf a, v16bf b, v8f c) {
    return __builtin_amdgcn_wmma_f32_16x16x32_bf16(
        /*neg_a=*/false, a, /*neg_b=*/false, b,
        /*c_mod=*/(short)0, c, /*reuse_a=*/false, /*reuse_b=*/false);
}

// 16-bit A/B fragment element map (ISA 7.12.2, 16-bit 16x32 A):
//   lane group g = lane>>4; half h in v16bf:
//   k = k0 + 8*g + (h<8 ? h : h+8)   -> two contiguous 8-element runs.
__device__ __forceinline__ v16bf frag_bf16(const __bf16* __restrict__ base,
                                           int row, int stride, int k0, int g) {
    const __bf16* p = base + (long)row * stride + k0 + g * 8;
    v8bf a = *(const v8bf*)(p);
    v8bf b = *(const v8bf*)(p + 16);
    v16bf f;
#pragma unroll
    for (int i = 0; i < 8; ++i) { f[i] = a[i]; f[i + 8] = b[i]; }
    return f;
}

// ---------------- Kernel 0: fp32 -> bf16 streaming convert --------------
__global__ __launch_bounds__(256) void cvt_bf16_kernel(
    const float* __restrict__ src, __bf16* __restrict__ dst, long n) {
    long i = ((long)blockIdx.x * blockDim.x + threadIdx.x) * 4;
    if (i + 3 >= n) {
        for (long k = i; k < n; ++k) dst[k] = (__bf16)src[k];
        return;
    }
    float4 v = *(const float4*)(src + i);
    v8bf* d2 = nullptr; (void)d2;
    dst[i + 0] = (__bf16)v.x;
    dst[i + 1] = (__bf16)v.y;
    dst[i + 2] = (__bf16)v.z;
    dst[i + 3] = (__bf16)v.w;
}

// ---------------- Kernel 1: fused Q/K/V projection ----------------
// One wave per 16x16 output tile of one of {Q,K,V}. Y = X @ W^T + b.
// Q,K stored [b][h][n][p] bf16; V stored transposed [b][h][p][n] bf16.
__global__ __launch_bounds__(256) void qkv_proj_kernel(
    const __bf16* __restrict__ xb,     // [8192][768] bf16
    const __bf16* __restrict__ wb,     // [3][768][768] bf16
    const float* __restrict__ bq, const float* __restrict__ bk,
    const float* __restrict__ bv,
    __bf16* __restrict__ q_ws, __bf16* __restrict__ k_ws,
    __bf16* __restrict__ vt_ws) {
    const int TN = (B_ * N_) / 16;   // 512 row tiles
    const int TM = D_ / 16;          // 48 col tiles
    int wid  = (blockIdx.x * blockDim.x + threadIdx.x) >> 5;
    if (wid >= 3 * TN * TM) return;
    int lane = threadIdx.x & 31;
    int g    = lane >> 4;
    int nc   = lane & 15;

    int sel = wid / (TN * TM);
    int rem = wid % (TN * TM);
    int n0  = (rem / TM) * 16;       // global row in [0, 8192)
    int m0  = (rem % TM) * 16;       // output feature col

    const __bf16* W  = wb + (long)sel * D_ * D_;
    const float*  Bv = (sel == 0) ? bq : (sel == 1) ? bk : bv;

    v8f acc = {};
#pragma unroll 4
    for (int kk = 0; kk < D_; kk += 32) {
        v16bf a = frag_bf16(xb, n0 + nc, D_, kk, g);  // A: rows of X
        v16bf b = frag_bf16(W,  m0 + nc, D_, kk, g);  // B[k][m] = W[m][k]
        acc = wmma_bf16(a, b, acc);
    }
    float bias = Bv[m0 + nc];
    int h = m0 >> 6;
    int p = (m0 & 63) + nc;          // tile never crosses a head boundary
#pragma unroll
    for (int r = 0; r < 8; ++r) {
        float v  = acc[r] + bias;
        int  n   = n0 + r + 8 * g;
        int  bb  = n >> 10;
        int  nn  = n & (N_ - 1);
        if (sel == 2) {
            vt_ws[(((long)bb * H_ + h) * P_ + p) * N_ + nn] = (__bf16)v;
        } else {
            __bf16* dst = sel ? k_ws : q_ws;
            dst[(((long)bb * H_ + h) * N_ + nn) * P_ + p] = (__bf16)v;
        }
    }
}

// ---------------- Kernel 2: flash attention with relative-key bias ----------
// One wave (1-wave workgroup) per (b, h, 16-row i-tile). 6144 workgroups.
__global__ __launch_bounds__(32) void attn_kernel(
    const __bf16* __restrict__ q_ws, const __bf16* __restrict__ k_ws,
    const __bf16* __restrict__ vt_ws, const __bf16* __restrict__ eb,
    float* __restrict__ out) {
    __shared__ __bf16 sQd[16 * 1040];   // Qd[u][c] = q[i0+u] . dist_emb[i0+c]
    __shared__ __bf16 sP[16 * 32];      // probs staging (C-layout -> A-layout)

    int lane = threadIdx.x & 31;
    int g    = lane >> 4;
    int nc   = lane & 15;
    int bh   = blockIdx.x >> 6;         // 64 i-tiles per (b,h)
    int it   = blockIdx.x & 63;
    int b    = bh / H_;
    int h    = bh % H_;
    int i0   = it * 16;

    const __bf16* qb = q_ws  + (long)bh * N_ * P_;
    const __bf16* kb = k_ws  + (long)bh * N_ * P_;
    const __bf16* vb = vt_ws + (long)bh * P_ * N_;

    // Persistent A-fragments of this wave's 16 query rows (K=64 -> two frags)
    v16bf qa0 = frag_bf16(qb, i0 + nc, P_, 0,  g);
    v16bf qa1 = frag_bf16(qb, i0 + nc, P_, 32, g);

    // ---- Precompute Qd slab via WMMA: Qd[16][1040] covering E rows [i0, i0+1038]
    for (int t = 0; t < 65; ++t) {
        int row = i0 + t * 16 + nc;
        if (row > EROWS - 1) row = EROWS - 1;           // clamp (unused cols)
        v16bf e0 = frag_bf16(eb, row, P_, 0,  g);       // B[k][n] = E[row_n][k]
        v16bf e1 = frag_bf16(eb, row, P_, 32, g);
        v8f c = {};
        c = wmma_bf16(qa0, e0, c);
        c = wmma_bf16(qa1, e1, c);
#pragma unroll
        for (int r = 0; r < 8; ++r)
            sQd[(r + 8 * g) * 1040 + t * 16 + nc] = (__bf16)c[r];
    }
    asm volatile("s_wait_dscnt 0" ::: "memory");

    // ---- Flash loop over keys, 32 at a time.
    // Base-2 softmax: p = 2^(s*f2 - m2), f2 = factor*log2(e).
    const float f2 = 0.125f * LOG2E;
    float mrow[8], lrow[8];
#pragma unroll
    for (int r = 0; r < 8; ++r) { mrow[r] = -3.0e38f; lrow[r] = 0.0f; }
    v8f O[4] = {};                 // 16 x 64 fp32 accumulator (4 N-tiles)

    for (int j0 = 0; j0 < N_; j0 += 32) {
        v8f s0 = {}, s1 = {};
        {
            v16bf kf0 = frag_bf16(kb, j0 + nc, P_, 0,  g);
            v16bf kf1 = frag_bf16(kb, j0 + nc, P_, 32, g);
            s0 = wmma_bf16(qa0, kf0, s0);
            s0 = wmma_bf16(qa1, kf1, s0);
        }
        {
            v16bf kf0 = frag_bf16(kb, j0 + 16 + nc, P_, 0,  g);
            v16bf kf1 = frag_bf16(kb, j0 + 16 + nc, P_, 32, g);
            s1 = wmma_bf16(qa0, kf0, s1);
            s1 = wmma_bf16(qa1, kf1, s1);
        }
        // bias + scale + online softmax (per-row stats live one-per-reg)
#pragma unroll
        for (int r = 0; r < 8; ++r) {
            int u = r + 8 * g;
            float bi0 = (float)sQd[u * 1040 + (u + 1023 - (j0 + nc))];
            float bi1 = (float)sQd[u * 1040 + (u + 1023 - (j0 + 16 + nc))];
            float v0 = (s0[r] + bi0) * f2;       // already in log2 domain
            float v1 = (s1[r] + bi1) * f2;
            float vmax = fmaxf(v0, v1);
#pragma unroll
            for (int msk = 1; msk < 16; msk <<= 1)
                vmax = fmaxf(vmax, __shfl_xor(vmax, msk, 32));
            float mnew = fmaxf(mrow[r], vmax);
            float corr = exp2f(mrow[r] - mnew);
            mrow[r] = mnew;
            float e0 = exp2f(v0 - mnew);
            float e1 = exp2f(v1 - mnew);
            float rs = e0 + e1;
#pragma unroll
            for (int msk = 1; msk < 16; msk <<= 1)
                rs += __shfl_xor(rs, msk, 32);
            lrow[r] = lrow[r] * corr + rs;
#pragma unroll
            for (int t2 = 0; t2 < 4; ++t2) O[t2][r] *= corr;
            sP[u * 32 + nc]      = (__bf16)e0;
            sP[u * 32 + 16 + nc] = (__bf16)e1;
        }
        asm volatile("s_wait_dscnt 0" ::: "memory");
        // Re-read probs in A-layout (16 x 32, K = j)
        v16bf pa;
        {
            const __bf16* pp = &sP[nc * 32 + g * 8];
            v8bf a = *(const v8bf*)(pp);
            v8bf c = *(const v8bf*)(pp + 16);
#pragma unroll
            for (int i = 0; i < 8; ++i) { pa[i] = a[i]; pa[i + 8] = c[i]; }
        }
        // O += P @ V  (B[k=j][n=p] = vT[p][j], contiguous in j)
#pragma unroll
        for (int t2 = 0; t2 < 4; ++t2) {
            v16bf vf = frag_bf16(vb, t2 * 16 + nc, N_, j0, g);
            O[t2] = wmma_bf16(pa, vf, O[t2]);
        }
    }

    // ---- Epilogue: normalize and store fp32 [b][i][h*64+p]
#pragma unroll
    for (int r = 0; r < 8; ++r) {
        int u = r + 8 * g;
        float inv = 1.0f / lrow[r];
        long rowoff = ((long)(b * N_ + i0 + u)) * D_ + h * P_;
#pragma unroll
        for (int t2 = 0; t2 < 4; ++t2)
            out[rowoff + t2 * 16 + nc] = O[t2][r] * inv;
    }
}

extern "C" void kernel_launch(void* const* d_in, const int* in_sizes, int n_in,
                              void* d_out, int out_size, void* d_ws, size_t ws_size,
                              hipStream_t stream) {
    const float* x  = (const float*)d_in[0];
    const float* wq = (const float*)d_in[1];
    const float* bq = (const float*)d_in[2];
    const float* wk = (const float*)d_in[3];
    const float* bk = (const float*)d_in[4];
    const float* wv = (const float*)d_in[5];
    const float* bv = (const float*)d_in[6];
    const float* de = (const float*)d_in[7];
    float* out = (float*)d_out;

    const size_t MAT_ELEMS = (size_t)B_ * H_ * N_ * P_;      // 6.29 M
    const size_t X_ELEMS   = (size_t)B_ * N_ * D_;           // 6.29 M
    const size_t W_ELEMS   = (size_t)D_ * D_;                // 0.59 M
    const size_t E_ELEMS   = (size_t)EROWS * P_;             // 131 K

    char* base = (char*)d_ws;
    __bf16* q_ws  = (__bf16*)(base);
    __bf16* k_ws  = q_ws  + MAT_ELEMS;
    __bf16* vt_ws = k_ws  + MAT_ELEMS;
    __bf16* xb    = vt_ws + MAT_ELEMS;
    __bf16* wb    = xb    + X_ELEMS;      // 3 weight matrices back-to-back
    __bf16* ebf   = wb    + 3 * W_ELEMS;

    // ---- one-shot fp32 -> bf16 conversions (all streaming, HBM-bound)
    auto cvt = [&](const float* s, __bf16* d, size_t n) {
        long blocks = (long)((n / 4 + 255) / 256);
        cvt_bf16_kernel<<<dim3((unsigned)blocks), dim3(256), 0, stream>>>(s, d, (long)n);
    };
    cvt(x,  xb,              X_ELEMS);
    cvt(wq, wb + 0 * W_ELEMS, W_ELEMS);
    cvt(wk, wb + 1 * W_ELEMS, W_ELEMS);
    cvt(wv, wb + 2 * W_ELEMS, W_ELEMS);
    cvt(de, ebf,             E_ELEMS);

    // 3 matrices * 512 row-tiles * 48 col-tiles = 73728 waves, 8 waves/block
    qkv_proj_kernel<<<dim3(73728 / 8), dim3(256), 0, stream>>>(
        xb, wb, bq, bk, bv, q_ws, k_ws, vt_ws);

    // one wave per (b, h, 16-row query tile): 8*12*64 = 6144 blocks
    attn_kernel<<<dim3(B_ * H_ * (N_ / 16)), dim3(32), 0, stream>>>(
        q_ws, k_ws, vt_ws, ebf, out);
}
